// MergedConv2d_57715770524152
// MI455X (gfx1250) — matrix operand order repaired and checked by reading, not compiled, not run on hardware
//
#include <hip/hip_runtime.h>

typedef __attribute__((ext_vector_type(16))) _Float16 v16h;
typedef __attribute__((ext_vector_type(8)))  float    v8f;

#define HH      320
#define WW      320
#define CIN_G   16
#define COUT_G  16
#define NGROUPS 4
#define LDSW    356                    // WW + 2*dmax (dmax = 18)
#define NWAVES  4
#define NTHREADS (NWAVES * 32)
#define ZOFF    (3 * LDSW * CIN_G)     // zeroed pad fragment for tap 9

// One workgroup = one (batch b, group g, output row y).
// Implicit GEMM: D[16 oc x 16 cols] = sum_{K=0..159} A[oc][K] * B[K][col],
// K = tap*16 + ic (tap 9 -> zero pad), via 5x v_wmma_f32_16x16x32_f16.
__global__ __launch_bounds__(NTHREADS)
void merged_dilated_conv_kernel(const float* __restrict__ x,
                                const float* __restrict__ wgt,
                                const float* __restrict__ bias,
                                float* __restrict__ out)
{
    const int y = blockIdx.x;        // output row
    const int g = blockIdx.y;        // group
    const int b = blockIdx.z;        // batch

    const int dil_tab[NGROUPS] = {1, 6, 12, 18};
    const int d = dil_tab[g];

    // LDS: 3 input rows (y-d, y, y+d) x (WW+2d) cols x 16 ic, f16,
    // layout [row][col][ic] -> one B fragment (16 ic) = 32 contiguous bytes.
    // Plus one zeroed fragment at ZOFF for the K-pad tap.
    __shared__ __align__(32) _Float16 lds[3 * LDSW * CIN_G + CIN_G];

    // ------------- stage: interior, 3 rows x 80 groups of 4 cols ----------
    // Each thread: 16 aligned float4 global loads (coalesced across wave),
    // f32->f16 via cvt_pk, then 4 conflict-free 32B LDS fragment stores.
    for (int q = threadIdx.x; q < 3 * (WW / 4); q += NTHREADS) {
        const int r    = q / (WW / 4);           // 0,1,2 -> rows y-d, y, y+d
        const int grp  = q - r * (WW / 4);
        const int ix0  = grp * 4;                // 16B-aligned column group
        const int colb = ix0 + d;                // LDS column base
        const int iy   = y + (r - 1) * d;
        v16h hb[4] = {};
        if (iy >= 0 && iy < HH) {
            const float* src =
                x + (((size_t)b * 64 + (size_t)g * CIN_G) * HH + iy) * WW + ix0;
            #pragma unroll
            for (int ic = 0; ic < CIN_G; ++ic) {
                const float4 fv = *(const float4*)(src + (size_t)ic * HH * WW);
                hb[0][ic] = (_Float16)fv.x;
                hb[1][ic] = (_Float16)fv.y;
                hb[2][ic] = (_Float16)fv.z;
                hb[3][ic] = (_Float16)fv.w;
            }
        }
        #pragma unroll
        for (int j = 0; j < 4; ++j)
            *(v16h*)&lds[(r * LDSW + colb + j) * CIN_G] = hb[j];
    }

    // ------------- stage: zero-pad columns (ix < 0 or ix >= WW) -----------
    {
        const v16h z = {};
        #pragma unroll
        for (int r = 0; r < 3; ++r)
            for (int o = threadIdx.x; o < 2 * d; o += NTHREADS) {
                const int col = (o < d) ? o : (WW + o);   // [0,d) U [WW+d,WW+2d)
                *(v16h*)&lds[(r * LDSW + col) * CIN_G] = z;
            }
        if (threadIdx.x == 0)
            *(v16h*)&lds[ZOFF] = z;                        // tap-9 pad fragment
    }

    // ------------- per-wave A (weight) fragments in registers -------------
    const int lane = threadIdx.x & 31;
    const int wave = threadIdx.x >> 5;
    const int hi   = lane >> 4;                  // half-wave id
    const int m    = lane & 15;                  // M (oc) / N (col) index

    // A 16x32 layout (matches B): lanes 0-15 hold K 0-15;
    // lanes 16-31 hold K 16-31 (each +i*32 per fragment).
    v16h a[5];
    #pragma unroll
    for (int i = 0; i < 5; ++i) {
        #pragma unroll
        for (int h = 0; h < 16; ++h) {
            const int k   = i * 32 + h + hi * 16;
            const int tap = k >> 4;              // 0..9 (9 = zero pad)
            const int ic  = k & 15;
            _Float16 wv = (_Float16)0.0f;
            if (tap < 9)
                wv = (_Float16)wgt[((g * COUT_G + m) * CIN_G + ic) * 9 + tap];
            a[i][h] = wv;
        }
    }

    float bv[8];
    #pragma unroll
    for (int v = 0; v < 8; ++v)
        bv[v] = bias[g * COUT_G + v + hi * 8];

    __syncthreads();

    // ------------- 20 column tiles of 16, round-robin over waves ----------
    for (int tile = wave; tile < WW / 16; tile += NWAVES) {
        const int x0 = tile * 16;
        v8f acc = {};
        #pragma unroll
        for (int i = 0; i < 5; ++i) {
            // B (32x16 f16): lanes 0-15 carry K=i*32+0..15 (tap 2i),
            // lanes 16-31 carry K=i*32+16..31 (tap 2i+1); halves = ic.
            const int tap = 2 * i + hi;
            const int ky  = tap / 3;
            const int kx  = tap - ky * 3;
            int off = (ky * LDSW + (x0 + m + kx * d)) * CIN_G;
            off = (tap <= 8) ? off : ZOFF;       // cndmask, no EXEC divergence
            const v16h bf = *(const v16h*)&lds[off];
            acc = __builtin_amdgcn_wmma_f32_16x16x32_f16(
                      false, a[i], false, bf, (short)0, acc, false, false);
        }
        // C/D layout: VGPR v -> M = v + 8*hi, N = lane&15
        #pragma unroll
        for (int v = 0; v < 8; ++v) {
            const int oc = v + hi * 8;
            out[(((size_t)b * 64 + (size_t)g * COUT_G + oc) * HH + y) * WW + x0 + m]
                = acc[v] + bv[v];
        }
    }
}

extern "C" void kernel_launch(void* const* d_in, const int* in_sizes, int n_in,
                              void* d_out, int out_size, void* d_ws, size_t ws_size,
                              hipStream_t stream) {
    const float* x    = (const float*)d_in[0];   // [8,64,320,320]
    const float* wgt  = (const float*)d_in[1];   // [64,16,3,3]
    const float* bias = (const float*)d_in[2];   // [64]
    float* out        = (float*)d_out;           // [8,64,320,320]

    dim3 grid(HH, NGROUPS, 8);                   // (y, group, batch)
    dim3 block(NTHREADS);
    merged_dilated_conv_kernel<<<grid, block, 0, stream>>>(x, wgt, bias, out);
}
